// NonLinearTransitionModel_45354854646239
// MI455X (gfx1250) — compile-verified, matrix operand order checked by reading, not run on hardware
//
#include <hip/hip_runtime.h>
#include <hip/hip_bf16.h>
#include <math.h>

// ---- problem constants (match reference) ----
#define NSTEPS 20
#define BROWS  8192
#define LATENT 256
#define UDIM   16
#define HIDDEN 512
#define KC1    9     // K chunks of 32 for GEMM1 (272 padded to 288)
#define NT1    32    // 512 / 16 output tiles, GEMM1
#define KC2    16    // 512 / 32 K chunks, GEMM2
#define NT2    16    // 256 / 16 output tiles, GEMM2
#define XPAD   296   // bf16 x-shadow row stride (592B, 16B aligned, bank step 20 -> conflict-free)
#define HPAD   520   // bf16 hidden row stride (1040B, bank step 4 -> conflict-free)
#define WGROWS 32    // rows per workgroup (two 16-row M tiles)
#define NWAVES 4     // waves per workgroup (n-split)

typedef __attribute__((ext_vector_type(16))) __bf16 bf16x16;
typedef __attribute__((ext_vector_type(8)))  __bf16 bf16x8;
typedef __attribute__((ext_vector_type(4)))  __bf16 bf16x4;
typedef __attribute__((ext_vector_type(8)))  float  f32x8;

#define WMMA_BF16(A, B, C) \
  __builtin_amdgcn_wmma_f32_16x16x32_bf16(false, (A), false, (B), (short)0, (C), false, false)

// gfx1250 has V_TANH_F32 (TRANS op); fall back to exp2/rcp identity (5 VALU ops).
__device__ __forceinline__ float fast_tanh(float x) {
#if __has_builtin(__builtin_amdgcn_tanhf)
  return __builtin_amdgcn_tanhf(x);
#else
  float t = __builtin_amdgcn_exp2f(x * 2.885390081777927f); // e^(2x)
  return 1.0f - 2.0f * __builtin_amdgcn_rcpf(t + 1.0f);
#endif
}

// Build a 16x32 bf16 A-fragment for this lane from a row-major LDS row:
// halfs 0..7 = K kb..kb+7, halfs 8..15 = K kb+16..kb+23 (two ds_load_b128).
__device__ __forceinline__ bf16x16 lds_afrag(const __bf16* p) {
  bf16x8 lo = *(const bf16x8*)p;
  bf16x8 hi = *(const bf16x8*)(p + 16);
  bf16x16 r;
  #pragma unroll
  for (int j = 0; j < 8; ++j) { r[j] = lo[j]; r[8 + j] = hi[j]; }
  return r;
}

// ---------------------------------------------------------------------------
// Weight packing: f32 row-major [K][N] -> bf16 WMMA B-fragments (ISA 7.12.2):
//   lane L holds column n = tile*16 + (L&15);
//   K base = chunk*32 + (L>=16 ? 8 : 0); halfs j<8 -> K=base+j, j>=8 -> K=base+16+(j-8)
// Fragment (tile*KC + chunk) = 512 contiguous bf16, lane offset lane*16.
// ---------------------------------------------------------------------------
__global__ void pack_w1_kernel(const float* __restrict__ W1, __bf16* __restrict__ w1p) {
  int idx = blockIdx.x * blockDim.x + threadIdx.x;
  if (idx >= KC1 * NT1 * 512) return;
  int frag = idx >> 9, within = idx & 511;
  int lane = within >> 4, j = within & 15;
  int t = frag / KC1, c = frag % KC1;
  int n  = t * 16 + (lane & 15);
  int kb = c * 32 + ((lane >= 16) ? 8 : 0);
  int k  = (j < 8) ? (kb + j) : (kb + 16 + (j - 8));
  float v = (k < (LATENT + UDIM)) ? W1[(size_t)k * HIDDEN + n] : 0.0f;
  w1p[idx] = (__bf16)v;
}

__global__ void pack_w2_kernel(const float* __restrict__ W2, __bf16* __restrict__ w2p) {
  int idx = blockIdx.x * blockDim.x + threadIdx.x;
  if (idx >= KC2 * NT2 * 512) return;
  int frag = idx >> 9, within = idx & 511;
  int lane = within >> 4, j = within & 15;
  int t = frag / KC2, c = frag % KC2;
  int n  = t * 16 + (lane & 15);
  int kb = c * 32 + ((lane >= 16) ? 8 : 0);
  int k  = (j < 8) ? (kb + j) : (kb + 16 + (j - 8));
  w2p[idx] = (__bf16)W2[(size_t)k * LATENT + n];
}

// ---------------------------------------------------------------------------
// Fused ODE kernel: 4-wave workgroup integrates 32 batch rows for all 20 steps.
//  - Two 16-row M tiles per workgroup: each B fragment feeds 2 WMMAs.
//  - n-split across the 4 waves keeps all 4 SIMD32s of a WGP busy.
//  - f32 z state lives in d_out (L2-resident, wave-private columns).
//  - Rolled k-loop processes chunk PAIRS (ping-pong cb/nb): the double-buffer
//    rotation is pure renaming across the two half-bodies -> no v_mov chains,
//    no WMMA WAR nop padding, fragments consumed 4-8 WMMAs after issue.
// ---------------------------------------------------------------------------
__global__ void __launch_bounds__(NWAVES * 32)
ode_fused_kernel(const float* __restrict__ zt, const float* __restrict__ dt,
                 const float* __restrict__ ut, const float* __restrict__ b1,
                 const float* __restrict__ b2,
                 const __bf16* __restrict__ w1p, const __bf16* __restrict__ w2p,
                 float* __restrict__ out)
{
  __shared__ __align__(16) __bf16 xbs[WGROWS][XPAD];  // bf16 shadow of x=[z;u;0]
  __shared__ __align__(16) __bf16 h1s[WGROWS][HPAD];  // tanh hidden tile, bf16
  __shared__ float hs[WGROWS];                        // per-row step size

  const int tid     = threadIdx.x;
  const int lane    = tid & 31;
  const int w       = tid >> 5;             // wave id 0..3
  const int rowBase = blockIdx.x * WGROWS;
  const int m       = lane & 15;            // A-fragment row (within a 16-row tile)
  const int khalf   = (lane >= 16) ? 8 : 0; // K sub-offset per lane half
  const int colq    = lane & 15;            // C-fragment column
  const int rhalf   = (lane >> 4) << 3;     // C-fragment row offset (0 or 8)

  // ---- init: wave w owns z columns [w*64, w*64+64) of all 32 rows ----
  {
    const int cbase = w * 64;
    #pragma unroll 4
    for (int i = 0; i < 16; ++i) {
      int linear = i * 32 + lane;           // 0..511
      int r  = linear >> 4;                 // 0..31
      int c4 = ((linear & 15) << 2) + cbase;
      float4 v = *(const float4*)&zt[(size_t)(rowBase + r) * LATENT + c4];
      *(float4*)&out[(size_t)(rowBase + r) * LATENT + c4] = v;  // f32 z state
      bf16x4 b; b[0] = (__bf16)v.x; b[1] = (__bf16)v.y; b[2] = (__bf16)v.z; b[3] = (__bf16)v.w;
      *(bf16x4*)&xbs[r][c4] = b;
    }
  }
  if (tid < WGROWS) {
    hs[tid] = dt[rowBase + tid] * (1.0f / NSTEPS);
    #pragma unroll
    for (int j = 0; j < UDIM; ++j)
      xbs[tid][LATENT + j] = (__bf16)ut[(size_t)(rowBase + tid) * UDIM + j];
    #pragma unroll
    for (int j = 0; j < 16; ++j)
      xbs[tid][LATENT + UDIM + j] = (__bf16)0.0f;   // K 272..287 zero pad
  }
  __syncthreads();

  // per-lane step sizes for the 8 C rows of each M tile (loop-invariant)
  float hv0[8], hv1[8];
  #pragma unroll
  for (int r = 0; r < 8; ++r) { hv0[r] = hs[rhalf + r]; hv1[r] = hs[16 + rhalf + r]; }

  const __bf16* w1l = w1p + (lane << 4);
  const __bf16* w2l = w2p + (lane << 4);
  const __bf16* xrow0 = &xbs[m][khalf];
  const __bf16* xrow1 = &xbs[16 + m][khalf];
  const __bf16* hrow0 = &h1s[m][khalf];
  const __bf16* hrow1 = &h1s[16 + m][khalf];

  #pragma unroll 1
  for (int step = 0; step < NSTEPS; ++step) {
    // ==== GEMM1: h1 = tanh(x @ W1 + b1). Wave w: n-tiles w*8..w*8+7 ====
    #pragma unroll 1
    for (int g = 0; g < 4; ++g) {
      const int t0 = w * 8 + g * 2, t1 = t0 + 1;
      const __bf16* p0 = w1l + ((size_t)(t0 * KC1) << 9);
      const __bf16* p1 = w1l + ((size_t)(t1 * KC1) << 9);
      bf16x16 cb0 = *(const bf16x16*)p0; p0 += 512;   // chunk 0
      bf16x16 cb1 = *(const bf16x16*)p1; p1 += 512;
      if (g < 3) {  // warm next group's first chunk (global_prefetch_b8)
        __builtin_prefetch(w1l + ((size_t)((t0 + 2) * KC1) << 9), 0, 3);
        __builtin_prefetch(w1l + ((size_t)((t1 + 2) * KC1) << 9), 0, 3);
      }
      f32x8 a00 = {}, a01 = {}, a10 = {}, a11 = {};   // [m-tile][n-tile]
      #pragma unroll 1
      for (int i = 0; i < (KC1 - 1) / 2; ++i) {       // chunk pairs (2i, 2i+1)
        const int c0 = 2 * i, c1 = 2 * i + 1;
        bf16x16 nb0 = *(const bf16x16*)p0; p0 += 512; // chunk c1
        bf16x16 nb1 = *(const bf16x16*)p1; p1 += 512;
        bf16x16 ae0 = lds_afrag(xrow0 + c0 * 32);
        bf16x16 ae1 = lds_afrag(xrow1 + c0 * 32);
        a00 = WMMA_BF16(ae0, cb0, a00);
        a10 = WMMA_BF16(ae1, cb0, a10);
        a01 = WMMA_BF16(ae0, cb1, a01);
        a11 = WMMA_BF16(ae1, cb1, a11);
        cb0 = *(const bf16x16*)p0; p0 += 512;         // chunk c1+1
        cb1 = *(const bf16x16*)p1; p1 += 512;
        bf16x16 ao0 = lds_afrag(xrow0 + c1 * 32);
        bf16x16 ao1 = lds_afrag(xrow1 + c1 * 32);
        a00 = WMMA_BF16(ao0, nb0, a00);
        a10 = WMMA_BF16(ao1, nb0, a10);
        a01 = WMMA_BF16(ao0, nb1, a01);
        a11 = WMMA_BF16(ao1, nb1, a11);
      }
      { // tail: chunk KC1-1 = 8, already in cb
        bf16x16 af0 = lds_afrag(xrow0 + (KC1 - 1) * 32);
        bf16x16 af1 = lds_afrag(xrow1 + (KC1 - 1) * 32);
        a00 = WMMA_BF16(af0, cb0, a00);
        a10 = WMMA_BF16(af1, cb0, a10);
        a01 = WMMA_BF16(af0, cb1, a01);
        a11 = WMMA_BF16(af1, cb1, a11);
      }
      // epilogue: bias + tanh -> h1s (wave-private columns)
      #pragma unroll
      for (int q = 0; q < 2; ++q) {
        f32x8 am0 = (q == 0) ? a00 : a01;
        f32x8 am1 = (q == 0) ? a10 : a11;
        int n0 = (q == 0 ? t0 : t1) * 16;
        float bv = b1[n0 + colq];
        #pragma unroll
        for (int r = 0; r < 8; ++r) {
          h1s[rhalf + r][n0 + colq]      = (__bf16)fast_tanh(am0[r] + bv);
          h1s[16 + rhalf + r][n0 + colq] = (__bf16)fast_tanh(am1[r] + bv);
        }
      }
    }
    __syncthreads();   // h1 complete before any wave consumes full K

    // ==== GEMM2: z += h * (h1 @ W2 + b2). Wave w: n-tiles w*4..w*4+3 ====
    #pragma unroll 1
    for (int g = 0; g < 2; ++g) {
      const int t0 = w * 4 + g * 2, t1 = t0 + 1;
      const __bf16* p0 = w2l + ((size_t)(t0 * KC2) << 9);
      const __bf16* p1 = w2l + ((size_t)(t1 * KC2) << 9);
      bf16x16 cb0 = *(const bf16x16*)p0; p0 += 512;   // chunk 0
      bf16x16 cb1 = *(const bf16x16*)p1; p1 += 512;
      if (g < 1) {
        __builtin_prefetch(w2l + ((size_t)((t0 + 2) * KC2) << 9), 0, 3);
        __builtin_prefetch(w2l + ((size_t)((t1 + 2) * KC2) << 9), 0, 3);
      }
      f32x8 a00 = {}, a01 = {}, a10 = {}, a11 = {};
      #pragma unroll 1
      for (int i = 0; i < (KC2 - 1) / 2; ++i) {       // chunk pairs (2i, 2i+1)
        const int c0 = 2 * i, c1 = 2 * i + 1;
        bf16x16 nb0 = *(const bf16x16*)p0; p0 += 512;
        bf16x16 nb1 = *(const bf16x16*)p1; p1 += 512;
        bf16x16 ae0 = lds_afrag(hrow0 + c0 * 32);
        bf16x16 ae1 = lds_afrag(hrow1 + c0 * 32);
        a00 = WMMA_BF16(ae0, cb0, a00);
        a10 = WMMA_BF16(ae1, cb0, a10);
        a01 = WMMA_BF16(ae0, cb1, a01);
        a11 = WMMA_BF16(ae1, cb1, a11);
        cb0 = *(const bf16x16*)p0; p0 += 512;
        cb1 = *(const bf16x16*)p1; p1 += 512;
        bf16x16 ao0 = lds_afrag(hrow0 + c1 * 32);
        bf16x16 ao1 = lds_afrag(hrow1 + c1 * 32);
        a00 = WMMA_BF16(ao0, nb0, a00);
        a10 = WMMA_BF16(ao1, nb0, a10);
        a01 = WMMA_BF16(ao0, nb1, a01);
        a11 = WMMA_BF16(ao1, nb1, a11);
      }
      { // tail: chunks 14 (in cb) and 15 (load into nb)
        bf16x16 nb0 = *(const bf16x16*)p0;
        bf16x16 nb1 = *(const bf16x16*)p1;
        bf16x16 ae0 = lds_afrag(hrow0 + (KC2 - 2) * 32);
        bf16x16 ae1 = lds_afrag(hrow1 + (KC2 - 2) * 32);
        a00 = WMMA_BF16(ae0, cb0, a00);
        a10 = WMMA_BF16(ae1, cb0, a10);
        a01 = WMMA_BF16(ae0, cb1, a01);
        a11 = WMMA_BF16(ae1, cb1, a11);
        bf16x16 ao0 = lds_afrag(hrow0 + (KC2 - 1) * 32);
        bf16x16 ao1 = lds_afrag(hrow1 + (KC2 - 1) * 32);
        a00 = WMMA_BF16(ao0, nb0, a00);
        a10 = WMMA_BF16(ao1, nb0, a10);
        a01 = WMMA_BF16(ao0, nb1, a01);
        a11 = WMMA_BF16(ao1, nb1, a11);
      }
      // epilogue: Euler update on f32 z state in d_out + refresh bf16 shadow
      #pragma unroll
      for (int q = 0; q < 2; ++q) {
        f32x8 am0 = (q == 0) ? a00 : a01;
        f32x8 am1 = (q == 0) ? a10 : a11;
        int n0 = (q == 0 ? t0 : t1) * 16;
        int col = n0 + colq;
        float bv = b2[col];
        #pragma unroll
        for (int r = 0; r < 8; ++r) {
          int rl0 = rhalf + r;
          size_t o0 = (size_t)(rowBase + rl0) * LATENT + col;
          float zn0 = out[o0] + hv0[r] * (am0[r] + bv);
          out[o0] = zn0;
          xbs[rl0][col] = (__bf16)zn0;
          int rl1 = 16 + rhalf + r;
          size_t o1 = (size_t)(rowBase + rl1) * LATENT + col;
          float zn1 = out[o1] + hv1[r] * (am1[r] + bv);
          out[o1] = zn1;
          xbs[rl1][col] = (__bf16)zn1;
        }
      }
    }
    __syncthreads();   // x shadow complete before next step's GEMM1
  }
  // z state already lives in d_out -- nothing further to write.
}

// ---------------------------------------------------------------------------
extern "C" void kernel_launch(void* const* d_in, const int* in_sizes, int n_in,
                              void* d_out, int out_size, void* d_ws, size_t ws_size,
                              hipStream_t stream) {
  const float* zt = (const float*)d_in[0];
  const float* dt = (const float*)d_in[1];
  const float* ut = (const float*)d_in[2];
  const float* W1 = (const float*)d_in[3];
  const float* b1 = (const float*)d_in[4];
  const float* W2 = (const float*)d_in[5];
  const float* b2 = (const float*)d_in[6];
  float* out = (float*)d_out;

  __bf16* w1p = (__bf16*)d_ws;                                   // 294,912 B
  __bf16* w2p = (__bf16*)((char*)d_ws + (size_t)KC1 * NT1 * 512 * sizeof(__bf16));

  const int n1 = KC1 * NT1 * 512;
  pack_w1_kernel<<<(n1 + 255) / 256, 256, 0, stream>>>(W1, w1p);
  const int n2 = KC2 * NT2 * 512;
  pack_w2_kernel<<<(n2 + 255) / 256, 256, 0, stream>>>(W2, w2p);

  dim3 grid(BROWS / WGROWS);   // 256 workgroups x 4 waves = 1024 waves
  ode_fused_kernel<<<grid, NWAVES * 32, 0, stream>>>(zt, dt, ut, b1, b2, w1p, w2p, out);
}
